// FeatureMatch_62938450756021
// MI455X (gfx1250) — compile-verified, matrix operand order checked by reading, not run on hardware
//
#include <hip/hip_runtime.h>

typedef _Float16 v8h  __attribute__((ext_vector_type(8)));
typedef _Float16 v16h __attribute__((ext_vector_type(16)));
typedef float    v8f  __attribute__((ext_vector_type(8)));

// ---------------------------------------------------------------------------
// Kernel 1: conv1 (3->64, 3x3 VALID) + ReLU + 2x2 maxpool, output f16.
// One thread per pooled output element. ~108 FMA/thread -> VALU is fine
// (only ~8% of total FLOPs).
// ---------------------------------------------------------------------------
__global__ void conv1_relu_pool(const float* __restrict__ in,
                                const float* __restrict__ w1,
                                const float* __restrict__ b1,
                                _Float16* __restrict__ out,
                                int B, int Hin, int Win, int Hp, int Wp) {
  int idx = blockIdx.x * blockDim.x + threadIdx.x;
  int total = B * 64 * Hp * Wp;
  if (idx >= total) return;
  int pw = idx % Wp; int t = idx / Wp;
  int ph = t % Hp;   t /= Hp;
  int oc = t % 64;   int b = t / 64;

  const float* wp = w1 + oc * 27;
  float bias = b1[oc];
  float best = -3.0e38f;
#pragma unroll
  for (int dy = 0; dy < 2; ++dy) {
#pragma unroll
    for (int dx = 0; dx < 2; ++dx) {
      int y0 = 2 * ph + dy;
      int x0 = 2 * pw + dx;
      float acc = bias;
#pragma unroll
      for (int ic = 0; ic < 3; ++ic) {
        const float* ip = in + ((size_t)(b * 3 + ic) * Hin + y0) * Win + x0;
#pragma unroll
        for (int r = 0; r < 3; ++r) {
#pragma unroll
          for (int s = 0; s < 3; ++s)
            acc += ip[(size_t)r * Win + s] * wp[ic * 9 + r * 3 + s];
        }
      }
      best = fmaxf(best, acc);
    }
  }
  best = fmaxf(best, 0.f);
  out[idx] = (_Float16)best;
}

// ---------------------------------------------------------------------------
// Kernel 2: repack conv2 weights [128][64][3][3] f32 -> f16 [M=128][K'=1024],
// 9 taps per input channel padded to 16 so one 16x16x32 WMMA K-step covers
// exactly two input channels with contiguous 128-bit A loads.
// ---------------------------------------------------------------------------
__global__ void prep_w2(const float* __restrict__ w2, _Float16* __restrict__ W2h) {
  int idx = blockIdx.x * blockDim.x + threadIdx.x;   // 0 .. 128*1024-1
  int kp = idx & 1023;
  int m  = idx >> 10;
  int ch = kp >> 4;
  int j  = kp & 15;
  _Float16 v = (_Float16)0.f;
  if (j < 9) v = (_Float16)w2[(size_t)(m * 64 + ch) * 9 + j];
  W2h[idx] = v;
}

// ---------------------------------------------------------------------------
// Kernel 3: conv2 (64->128, 3x3 VALID) as implicit GEMM via
// v_wmma_f32_16x16x32_f16, with bias + 2x2 maxpool + ReLU fused in the
// epilogue. Block = 256 threads = 8 waves; wave w owns M-tile of 16 output
// channels. N-tile = 16 positions laid out as 2 conv rows x 8 conv cols so
// pooling reduces across lanes with shuffles.
// B operand staged in LDS as ldsB[n][K'] (32KB) -> two ds_load_b128 per step.
// ---------------------------------------------------------------------------
__global__ void conv2_wmma_pool(const _Float16* __restrict__ F1,
                                const _Float16* __restrict__ W2h,
                                const float* __restrict__ b2,
                                float* __restrict__ out,
                                int Hin, int Win, int Hp, int Wp, int nct) {
  __shared__ __align__(16) _Float16 ldsB[16 * 1024];

  const int bz  = blockIdx.z;                 // batch
  const int ph  = blockIdx.x / nct;           // pooled row
  const int pcb = (blockIdx.x % nct) * 4;     // pooled col base (4 per tile)
  const int tid = threadIdx.x;

  // ---- stage im2col tile into LDS: ldsB[n][k'], n = rp*8 + colofs ----
  {
    int n      = tid & 15;
    int rp     = n >> 3;
    int colofs = n & 7;
    int oh = 2 * ph + rp;          // conv output row
    int ow = 2 * pcb + colofs;     // conv output col
    int k0 = (tid >> 4) * 64;      // 64 consecutive K' per thread
    const _Float16* base = F1 + (size_t)bz * 64 * Hin * Win;
#pragma unroll 4
    for (int kk = 0; kk < 64; ++kk) {
      int kp = k0 + kk;
      int ch = kp >> 4;
      int j  = kp & 15;
      _Float16 v = (_Float16)0.f;
      if (j < 9) {
        int r = j / 3, s = j - r * 3;
        int y = oh + r, x = ow + s;
        if (y < Hin && x < Win)
          v = base[((size_t)ch * Hin + y) * Win + x];
      }
      ldsB[n * 1024 + kp] = v;
    }
  }
  __syncthreads();

  const int wave  = tid >> 5;        // 0..7 -> M-tile
  const int lane  = tid & 31;
  const int hi    = lane >> 4;       // half-wave selector
  const int mbase = wave * 16;
  const int n     = lane & 15;

  const _Float16* arow = W2h + (size_t)(mbase + n) * 1024;
  const _Float16* brow = &ldsB[n * 1024];
  __builtin_prefetch(arow, 0, 3);

  v8f acc = {};
#pragma unroll 4
  for (int c2 = 0; c2 < 32; ++c2) {
    int ko = c2 * 32 + hi * 8;
    // A: weights, row-major -> two contiguous 128-bit loads per lane
    v8h a0 = *(const v8h*)(arow + ko);
    v8h a1 = *(const v8h*)(arow + ko + 16);
    v16h A = __builtin_shufflevector(a0, a1, 0,1,2,3,4,5,6,7,8,9,10,11,12,13,14,15);
    // B: activations from LDS, two ds_load_b128 per lane
    v8h c0 = *(const v8h*)(brow + ko);
    v8h c1 = *(const v8h*)(brow + ko + 16);
    v16h Bm = __builtin_shufflevector(c0, c1, 0,1,2,3,4,5,6,7,8,9,10,11,12,13,14,15);
    acc = __builtin_amdgcn_wmma_f32_16x16x32_f16(
        /*neg_a=*/false, A, /*neg_b=*/false, Bm,
        /*c_mod=*/(short)0, acc, /*reuse_a=*/false, /*reuse_b=*/false);
  }

  // ---- epilogue: bias, 2x2 maxpool via cross-lane max, ReLU, store ----
#pragma unroll
  for (int i = 0; i < 8; ++i) {
    int m = mbase + hi * 8 + i;           // C/D layout: VGPR i, half-wave hi
    float v = acc[i] + b2[m];
    v = fmaxf(v, __shfl_xor(v, 8, 32));   // max over row parity (n ^ 8)
    v = fmaxf(v, __shfl_xor(v, 1, 32));   // max over col pair   (n ^ 1)
    if ((n & 9) == 0) {                   // n in {0,2,4,6}: pooled writer
      int pc = pcb + (n >> 1);
      if (pc < Wp)
        out[(((size_t)bz * 128 + m) * Hp + ph) * Wp + pc] = fmaxf(v, 0.f);
    }
  }
}

// ---------------------------------------------------------------------------
// Kernel 4: depthwise cross-correlation. One block per (batch, channel);
// 62x62 feature map + 14x14 kernel staged in LDS, f32 FMAs.
// ---------------------------------------------------------------------------
__global__ void xcorr_dw(const float* __restrict__ X,
                         const float* __restrict__ K1,
                         const float* __restrict__ K2,
                         float* __restrict__ out) {
  __shared__ float Xs[62 * 62];
  __shared__ float Ks[196];
  int bc = blockIdx.x;        // (b*128 + c), 0..4095
  int tp = blockIdx.y;        // template 0/1
  int tid = threadIdx.x;

  const float* Kp = (tp == 0 ? K1 : K2) + (size_t)bc * 196;
  const float* Xp = X + (size_t)bc * (62 * 62);
  for (int i = tid; i < 62 * 62; i += 256) Xs[i] = Xp[i];
  if (tid < 196) Ks[tid] = Kp[tid];
  __syncthreads();

  float* op = out + (size_t)tp * 4096 * 2401 + (size_t)bc * 2401;
  for (int p = tid; p < 2401; p += 256) {
    int oy = p / 49, ox = p - oy * 49;
    float acc = 0.f;
#pragma unroll
    for (int r = 0; r < 14; ++r) {
      const float* xr = &Xs[(oy + r) * 62 + ox];
      const float* kr = &Ks[r * 14];
#pragma unroll
      for (int s = 0; s < 14; ++s) acc += xr[s] * kr[s];
    }
    op[p] = acc;
  }
}

// ---------------------------------------------------------------------------
// Host launcher
// ---------------------------------------------------------------------------
extern "C" void kernel_launch(void* const* d_in, const int* in_sizes, int n_in,
                              void* d_out, int out_size, void* d_ws, size_t ws_size,
                              hipStream_t stream) {
  (void)in_sizes; (void)n_in; (void)out_size; (void)ws_size;
  const float* img = (const float*)d_in[0];   // [32,3,256,256]
  const float* t1  = (const float*)d_in[1];   // [32,3,64,64]
  const float* t2  = (const float*)d_in[2];   // [32,3,64,64]
  const float* w1  = (const float*)d_in[3];   // [64,3,3,3]
  const float* b1  = (const float*)d_in[4];   // [64]
  const float* w2  = (const float*)d_in[5];   // [128,64,3,3]
  const float* b2  = (const float*)d_in[6];   // [128]
  float* out = (float*)d_out;                 // [2][32,128,49,49]

  // Workspace layout (all sections 256-B aligned)
  char* ws = (char*)d_ws;
  _Float16* F1img = (_Float16*)(ws + 0);             // 32*64*127*127 h = 66,064,384 B
  _Float16* F1t1  = (_Float16*)(ws + 66064384);      // 32*64*31*31  h =  3,936,256 B
  _Float16* F1t2  = (_Float16*)(ws + 69998592);      //                  3,936,256 B
  _Float16* W2h   = (_Float16*)(ws + 73932800);      // 128*1024 h   =     262,144 B
  float*    Xf    = (float*)   (ws + 74194944);      // 32*128*62*62 =  62,980,096 B
  float*    K1f   = (float*)   (ws + 137175040);     // 32*128*14*14 =   3,211,264 B
  float*    K2f   = (float*)   (ws + 140386304);     //                  3,211,264 B

  // 1) conv2 weight repack -> f16 padded-K layout
  prep_w2<<<dim3(512), dim3(256), 0, stream>>>(w2, W2h);

  // 2) conv1+relu+pool (f16 output)
  {
    int tot = 32 * 64 * 127 * 127;
    conv1_relu_pool<<<dim3((tot + 255) / 256), dim3(256), 0, stream>>>(
        img, w1, b1, F1img, 32, 256, 256, 127, 127);
  }
  {
    int tot = 32 * 64 * 31 * 31;
    conv1_relu_pool<<<dim3((tot + 255) / 256), dim3(256), 0, stream>>>(
        t1, w1, b1, F1t1, 32, 64, 64, 31, 31);
    conv1_relu_pool<<<dim3((tot + 255) / 256), dim3(256), 0, stream>>>(
        t2, w1, b1, F1t2, 32, 64, 64, 31, 31);
  }

  // 3) conv2 WMMA + bias + pool + relu
  //    image: Hin=Win=127, pooled 62x62, 16 col-tiles of 4 pooled cols
  conv2_wmma_pool<<<dim3(62 * 16, 1, 32), dim3(256), 0, stream>>>(
      F1img, W2h, b2, Xf, 127, 127, 62, 62, 16);
  //    templates: Hin=Win=31, pooled 14x14, 4 col-tiles
  conv2_wmma_pool<<<dim3(14 * 4, 1, 32), dim3(256), 0, stream>>>(
      F1t1, W2h, b2, K1f, 31, 31, 14, 14, 4);
  conv2_wmma_pool<<<dim3(14 * 4, 1, 32), dim3(256), 0, stream>>>(
      F1t2, W2h, b2, K2f, 31, 31, 14, 14, 4);

  // 4) depthwise cross-correlation for both templates
  xcorr_dw<<<dim3(4096, 2), dim3(256), 0, stream>>>(Xf, K1f, K2f, out);
}
